// GATEncoder_90056874262576
// MI455X (gfx1250) — matrix operand order repaired and checked by reading, not compiled, not run on hardware
//
#include <hip/hip_runtime.h>
#include <cstddef>

// ---------------------------------------------------------------------------
// GAT encoder for gfx1250 (MI455X). Dense matmuls on v_wmma_f32_16x16x32_bf16
// (wave32, f32 accumulate). A-tiles staged LDS-direct by the Tensor Data
// Mover (tensor_load_to_lds + s_wait_tensorcnt). Attention matrix lives as
// bf16 inside the A_pred region of d_out (134 MB < 192 MB L2).
// ---------------------------------------------------------------------------

typedef __attribute__((ext_vector_type(16))) __bf16        v16bf;
typedef __attribute__((ext_vector_type(8)))  float         v8f;
typedef __attribute__((ext_vector_type(4)))  unsigned int  u32x4;
typedef __attribute__((ext_vector_type(4)))  int           i32x4;
typedef __attribute__((ext_vector_type(8)))  int           i32x8;

#if defined(__has_builtin)
#if __has_builtin(__builtin_amdgcn_tensor_load_to_lds) && \
    __has_builtin(__builtin_amdgcn_s_wait_tensorcnt)
#define USE_TDM 1
#endif
#endif
#ifndef USE_TDM
#define USE_TDM 0
#endif

#define NN    8192
#define FIN   512
#define HID   512
#define EMB   128

#define BM 64
#define BN 64
#define BK 32
#define TSTR 40   // LDS row stride (ushorts): even, 80B = multiple of 16B

__device__ __forceinline__ unsigned short f2bf(float f) {
  union { float f; unsigned int u; } c; c.f = f;
  unsigned int u = c.u;
  unsigned int rb = 0x7fffu + ((u >> 16) & 1u);   // round-to-nearest-even
  return (unsigned short)((u + rb) >> 16);
}
__device__ __forceinline__ float bf2f(unsigned short s) {
  union { unsigned int u; float f; } c; c.u = ((unsigned int)s) << 16;
  return c.f;
}

// ---------------------------------------------------------------------------
// f32 -> bf16 conversion (grid-stride)
// ---------------------------------------------------------------------------
__global__ void cvt_bf16_kernel(const float* __restrict__ src,
                                unsigned short* __restrict__ dst, size_t n) {
  size_t i = (size_t)blockIdx.x * blockDim.x + threadIdx.x;
  size_t stride = (size_t)gridDim.x * blockDim.x;
  for (; i < n; i += stride) dst[i] = f2bf(src[i]);
}

// ---------------------------------------------------------------------------
// WMMA GEMM: C[M,N] = act(A[M,K] @ B[K,N])     (A,B bf16; C f32 and/or bf16)
// bTrans: B element (k,n) is at Bg[n*ldb + k] (for z @ z^T)
// act: 0 = none, 1 = ELU, 2 = sigmoid
// Requires M%64==0, N%64==0, K%32==0 (true for every call here).
// ---------------------------------------------------------------------------
__global__ __launch_bounds__(256)
void wmma_gemm_kernel(const unsigned short* __restrict__ A,
                      const unsigned short* __restrict__ B,
                      float* __restrict__ outF, unsigned short* __restrict__ outB,
                      int Mdim, int Ndim, int Kdim,
                      int lda, int ldb, int ldc, int bTrans, int act) {
  __shared__ alignas(16) unsigned short As [BM][TSTR];  // A tile, K-contiguous rows
  __shared__ alignas(16) unsigned short BsT[BN][TSTR];  // B tile, N-major (k contiguous)

  const int tid   = threadIdx.x;
  const int lane  = tid & 31;
  const int wave  = tid >> 5;          // 8 waves
  const int waveM = wave >> 1;         // 0..3 -> 16-row strip
  const int waveN = wave & 1;          // 0..1 -> 32-col strip
  const int half  = lane >> 4;         // fragment lane-half (ISA 7.12.2)
  const int l16   = lane & 15;
  const int bm0   = blockIdx.y * BM;
  const int bn0   = blockIdx.x * BN;

  v8f c0 = {}; v8f c1 = {};

  const int am = tid >> 2, aseg  = tid & 3;   // manual A staging: 64 x 4 x uint4
  const int bk = tid >> 3, bsegN = tid & 7;   // B staging: 32 k x 8 x uint4
  const int bn = tid >> 2, bsegT = tid & 3;   // B^T staging: 64 n x 4 x uint4

  for (int k0 = 0; k0 < Kdim; k0 += BK) {
#if USE_TDM
    if (wave == 0) {
      // Tensor DMA descriptor: 2D tile 32(K) x 64(M), 2B elements, LDS row
      // padded by 4 DWORDs every 16 DWORDs -> LDS stride 40 ushorts (= TSTR).
      const unsigned long long ga =
          (unsigned long long)(size_t)(A + (size_t)bm0 * lda + k0);
      const unsigned lds_addr = (unsigned)(size_t)(&As[0][0]); // flat low 32 = LDS addr
      const unsigned td0 = (unsigned)(Kdim - k0);
      const unsigned td1 = (unsigned)(Mdim - bm0);
      u32x4 g0;
      g0[0] = 1u;                                       // count=1, no gather
      g0[1] = lds_addr;
      g0[2] = (unsigned)(ga & 0xffffffffu);
      g0[3] = ((unsigned)(ga >> 32) & 0x01ffffffu) | (2u << 30); // type=2
      i32x8 g1;
      g1[0] = (int)0x06D10000u;   // data_size=2B | pad_en | interval=16DW | amount=4DW
      g1[1] = (int)(td0 << 16);                         // tensor_dim0 lo16
      g1[2] = (int)((td0 >> 16) | (td1 << 16));         // dim0 hi | dim1 lo
      g1[3] = (int)((td1 >> 16) | (32u << 16));         // dim1 hi | tile_dim0=32
      g1[4] = (int)64u;                                 // tile_dim1=64, tile_dim2=0
      g1[5] = (int)(unsigned)lda;                       // tensor_dim0_stride lo32
      g1[6] = 0;                                        // stride0 hi | stride1 lo
      g1[7] = 0;
      i32x4 gz = {0, 0, 0, 0};
#if __clang_major__ >= 23
      i32x8 gz8 = {0, 0, 0, 0, 0, 0, 0, 0};
      __builtin_amdgcn_tensor_load_to_lds(g0, g1, gz, gz, gz8, 0);
#else
      __builtin_amdgcn_tensor_load_to_lds(g0, g1, gz, gz, 0);
#endif
    }
#else
    { // manual A staging (BM x BK)
      uint4 v = *(const uint4*)(A + (size_t)(bm0 + am) * lda + (k0 + aseg * 8));
      *(uint4*)&As[am][aseg * 8] = v;
    }
#endif
    if (!bTrans) { // B row-major: vector global load, transposing b16 stores
      union { uint4 v; unsigned short s[8]; } u;
      u.v = *(const uint4*)(B + (size_t)(k0 + bk) * ldb + (bn0 + bsegN * 8));
#pragma unroll
      for (int i = 0; i < 8; ++i) BsT[bsegN * 8 + i][bk] = u.s[i];
    } else {       // B^T (z@z^T): k-contiguous in global, vector LDS store
      uint4 v = *(const uint4*)(B + (size_t)(bn0 + bn) * ldb + (k0 + bsegT * 8));
      *(uint4*)&BsT[bn][bsegT * 8] = v;
    }
#if USE_TDM
    if (wave == 0) __builtin_amdgcn_s_wait_tensorcnt(0);
#endif
    __syncthreads();

    // A fragment: lane = row waveM*16+l16; elems 0-15 = K kh..kh+15
    // contiguous -> two 16B LDS loads (80B row stride).
    const int ar = waveM * 16 + l16;
    const int kh = half * 16;
    union { uint4 q[2]; v16bf v; } ua;
    ua.q[0] = *(const uint4*)&As[ar][kh];
    ua.q[1] = *(const uint4*)&As[ar][kh + 8];
    // B fragments: lane = col; K kb..kb+15 contiguous in BsT row -> 32B.
    const int bc = waveN * 32 + l16;
    const int kb = half * 16;
    union { uint4 q[2]; v16bf v; } ub0, ub1;
    ub0.q[0] = *(const uint4*)&BsT[bc][kb];
    ub0.q[1] = *(const uint4*)&BsT[bc][kb + 8];
    ub1.q[0] = *(const uint4*)&BsT[bc + 16][kb];
    ub1.q[1] = *(const uint4*)&BsT[bc + 16][kb + 8];

    c0 = __builtin_amdgcn_wmma_f32_16x16x32_bf16(false, ua.v, false, ub0.v, (short)0, c0, false, false);
    c1 = __builtin_amdgcn_wmma_f32_16x16x32_bf16(false, ua.v, false, ub1.v, (short)0, c1, false, false);
    __syncthreads();
  }

  // D layout: VGPR v -> row v + 8*half, lane -> col l16
  const int rb = bm0 + waveM * 16 + half * 8;
  const int cb = bn0 + waveN * 32 + l16;
#pragma unroll
  for (int v = 0; v < 8; ++v) {
    int r = rb + v;
    if (r >= Mdim) continue;
    float x0 = c0[v], x1 = c1[v];
    if (act == 1) {        // ELU
      x0 = x0 > 0.f ? x0 : expm1f(x0);
      x1 = x1 > 0.f ? x1 : expm1f(x1);
    } else if (act == 2) { // sigmoid
      x0 = 1.f / (1.f + __expf(-x0));
      x1 = 1.f / (1.f + __expf(-x1));
    }
    if (outF) {
      if (cb < Ndim)      outF[(size_t)r * ldc + cb]      = x0;
      if (cb + 16 < Ndim) outF[(size_t)r * ldc + cb + 16] = x1;
    }
    if (outB) {
      if (cb < Ndim)      outB[(size_t)r * ldc + cb]      = f2bf(x0);
      if (cb + 16 < Ndim) outB[(size_t)r * ldc + cb + 16] = f2bf(x1);
    }
  }
}

// ---------------------------------------------------------------------------
// s[i] = h[i,:].a_self ; n[i] = h[i,:].a_neigh   (h bf16). 1 wave per row.
// ---------------------------------------------------------------------------
__global__ __launch_bounds__(256)
void rowdot_kernel(const unsigned short* __restrict__ h,
                   const float* __restrict__ a_self, const float* __restrict__ a_neigh,
                   float* __restrict__ s, float* __restrict__ n, int rows, int K) {
  int wave = threadIdx.x >> 5, lane = threadIdx.x & 31;
  int row = blockIdx.x * 8 + wave;
  if (row >= rows) return;
  float ss = 0.f, nn = 0.f;
  for (int k = lane; k < K; k += 32) {
    float hv = bf2f(h[(size_t)row * K + k]);
    ss += hv * a_self[k];
    nn += hv * a_neigh[k];
  }
#pragma unroll
  for (int o = 16; o; o >>= 1) {
    ss += __shfl_down(ss, o, 32);
    nn += __shfl_down(nn, o, 32);
  }
  if (lane == 0) { s[row] = ss; n[row] = nn; }
}

// ---------------------------------------------------------------------------
// Per-row masked softmax attention. One block per row; 32KB score row in LDS.
// attn[i,j] = softmax_j( mask(leaky((s_i + n_j) * M_ij)) ) stored as bf16.
// ---------------------------------------------------------------------------
__global__ __launch_bounds__(256)
void attn_row_kernel(const float* __restrict__ s, const float* __restrict__ nvec,
                     const float* __restrict__ Mm, const float* __restrict__ adj,
                     unsigned short* __restrict__ attn, int N) {
  __shared__ float sc[NN];
  __shared__ float red[256];
  const int i = blockIdx.x, tid = threadIdx.x;
  const float si = s[i];
  const size_t ro = (size_t)i * N;

  float mx = -3.0e38f;
  for (int j = tid; j < N; j += 256) {
    float e = (si + nvec[j]) * Mm[ro + j];
    e = e > 0.f ? e : 0.2f * e;                 // leaky relu (ALPHA=0.2)
    e = adj[ro + j] > 0.f ? e : -9.0e15f;       // mask
    sc[j] = e;
    mx = fmaxf(mx, e);
  }
  red[tid] = mx; __syncthreads();
#pragma unroll
  for (int o = 128; o; o >>= 1) { if (tid < o) red[tid] = fmaxf(red[tid], red[tid + o]); __syncthreads(); }
  mx = red[0];

  float sum = 0.f;
  for (int j = tid; j < N; j += 256) {
    float ex = __expf(sc[j] - mx);
    sc[j] = ex;
    sum += ex;
  }
  __syncthreads();              // everyone consumed red[0] before reuse
  red[tid] = sum; __syncthreads();
#pragma unroll
  for (int o = 128; o; o >>= 1) { if (tid < o) red[tid] += red[tid + o]; __syncthreads(); }
  const float inv = 1.f / red[0];

  for (int j = tid; j < N; j += 256) attn[ro + j] = f2bf(sc[j] * inv);
}

// ---------------------------------------------------------------------------
// Row L2-normalize: z (f32, into d_out tail) and zb (bf16, workspace).
// ---------------------------------------------------------------------------
__global__ __launch_bounds__(256)
void norm_rows_kernel(const float* __restrict__ h, float* __restrict__ z,
                      unsigned short* __restrict__ zb, int rows, int E) {
  int wave = threadIdx.x >> 5, lane = threadIdx.x & 31;
  int row = blockIdx.x * 8 + wave;
  if (row >= rows) return;
  float ss = 0.f;
  for (int k = lane; k < E; k += 32) { float v = h[(size_t)row * E + k]; ss += v * v; }
#pragma unroll
  for (int o = 16; o; o >>= 1) ss += __shfl_down(ss, o, 32);
  ss = __shfl(ss, 0, 32);
  float inv = 1.f / fmaxf(sqrtf(ss), 1e-12f);
  for (int k = lane; k < E; k += 32) {
    float v = h[(size_t)row * E + k] * inv;
    z[(size_t)row * E + k]  = v;
    zb[(size_t)row * E + k] = f2bf(v);
  }
}

// ---------------------------------------------------------------------------
extern "C" void kernel_launch(void* const* d_in, const int* in_sizes, int n_in,
                              void* d_out, int out_size, void* d_ws, size_t ws_size,
                              hipStream_t stream) {
  const float* x        = (const float*)d_in[0];
  const float* adj      = (const float*)d_in[1];
  const float* Mm       = (const float*)d_in[2];
  const float* W1       = (const float*)d_in[3];
  const float* a_self1  = (const float*)d_in[4];
  const float* a_neigh1 = (const float*)d_in[5];
  const float* W2       = (const float*)d_in[6];
  const float* a_self2  = (const float*)d_in[7];
  const float* a_neigh2 = (const float*)d_in[8];
  float* out = (float*)d_out;

  // workspace carve-out (256B-aligned); total ~33 MB
  char* ws = (char*)d_ws;
  size_t off = 0;
  auto carve = [&](size_t bytes) -> void* {
    void* p = ws + off; off += (bytes + 255) & ~(size_t)255; return p;
  };
  unsigned short* xb   = (unsigned short*)carve((size_t)NN * FIN * 2);
  unsigned short* W1b  = (unsigned short*)carve((size_t)FIN * HID * 2);
  unsigned short* W2b  = (unsigned short*)carve((size_t)HID * EMB * 2);
  unsigned short* h1b  = (unsigned short*)carve((size_t)NN * HID * 2);
  unsigned short* hp1b = (unsigned short*)carve((size_t)NN * HID * 2);
  unsigned short* h2b  = (unsigned short*)carve((size_t)NN * EMB * 2);
  float*          hp2f = (float*)         carve((size_t)NN * EMB * 4);
  float*          sv   = (float*)         carve((size_t)NN * 4);
  float*          nv   = (float*)         carve((size_t)NN * 4);
  unsigned short* zb   = (unsigned short*)carve((size_t)NN * EMB * 2);

  // bf16 attention matrix in the (dead-until-final-GEMM) A_pred region of
  // d_out: 134 MB bf16 inside 268 MB f32 -> L2-resident (192 MB L2).
  unsigned short* attn = (unsigned short*)d_out;
  float* z_out = out + (size_t)NN * NN;

  // --- convert inputs to bf16 ---
  cvt_bf16_kernel<<<4096, 256, 0, stream>>>(x,  xb,  (size_t)NN * FIN);
  cvt_bf16_kernel<<<1024, 256, 0, stream>>>(W1, W1b, (size_t)FIN * HID);
  cvt_bf16_kernel<<<256,  256, 0, stream>>>(W2, W2b, (size_t)HID * EMB);

  // --- layer 1 ---
  wmma_gemm_kernel<<<dim3(HID / 64, NN / 64), 256, 0, stream>>>(
      xb, W1b, nullptr, h1b, NN, HID, FIN, FIN, HID, HID, 0, 0);        // h = x@W1
  rowdot_kernel<<<NN / 8, 256, 0, stream>>>(h1b, a_self1, a_neigh1, sv, nv, NN, HID);
  attn_row_kernel<<<NN, 256, 0, stream>>>(sv, nv, Mm, adj, attn, NN);
  wmma_gemm_kernel<<<dim3(HID / 64, NN / 64), 256, 0, stream>>>(
      attn, h1b, nullptr, hp1b, NN, HID, NN, NN, HID, HID, 0, 1);       // elu(attn@h)

  // --- layer 2 ---
  wmma_gemm_kernel<<<dim3(EMB / 64, NN / 64), 256, 0, stream>>>(
      hp1b, W2b, nullptr, h2b, NN, EMB, HID, HID, EMB, EMB, 0, 0);      // h2 = h'@W2
  rowdot_kernel<<<NN / 8, 256, 0, stream>>>(h2b, a_self2, a_neigh2, sv, nv, NN, EMB);
  attn_row_kernel<<<NN, 256, 0, stream>>>(sv, nv, Mm, adj, attn, NN);
  wmma_gemm_kernel<<<dim3(EMB / 64, NN / 64), 256, 0, stream>>>(
      attn, h2b, hp2f, nullptr, NN, EMB, NN, NN, EMB, EMB, 0, 1);       // elu(attn@h2)

  // --- decoder ---
  norm_rows_kernel<<<NN / 8, 256, 0, stream>>>(hp2f, z_out, zb, NN, EMB);
  wmma_gemm_kernel<<<dim3(NN / 64, NN / 64), 256, 0, stream>>>(
      zb, zb, out, nullptr, NN, NN, EMB, EMB, EMB, NN, 1, 2);           // sigmoid(z@z^T)
}
